// EGCN_48189533061713
// MI455X (gfx1250) — compile-verified, matrix operand order checked by reading, not run on hardware
//
#include <hip/hip_runtime.h>
#include <hip/hip_bf16.h>

// CDNA5 / gfx1250: wave32, WMMA 16x16x32 f16 path.
typedef __attribute__((ext_vector_type(16))) _Float16 v16h;
typedef __attribute__((ext_vector_type(8)))  float    v8f;

#define N_NODES 50000
#define E_EDGES 600000
#define T_STEPS 6
#define DIN     128
#define DFF     2048
#define LN_EPS  1e-6f

// ---------------------------------------------------------------- utilities
__global__ void k_zero_f32(float* p, long n) {
  long i = (long)blockIdx.x * blockDim.x + threadIdx.x;
  long s = (long)gridDim.x * blockDim.x;
  for (; i < n; i += s) p[i] = 0.f;
}

__global__ void k_relu(float* p, long n) {
  long i = (long)blockIdx.x * blockDim.x + threadIdx.x;
  long s = (long)gridDim.x * blockDim.x;
  for (; i < n; i += s) p[i] = fmaxf(p[i], 0.f);
}

__global__ void k_zero_i32(int* p, int n) {
  int i = blockIdx.x * blockDim.x + threadIdx.x;
  if (i < n) p[i] = 0;
}

// 1/||scorer||_2 with a single wave32
__global__ void k_inv_norm(const float* v, int n, float* out) {
  float s = 0.f;
  for (int i = threadIdx.x; i < n; i += 32) s += v[i] * v[i];
  for (int o = 16; o; o >>= 1) s += __shfl_xor(s, o);
  if (threadIdx.x == 0) *out = rsqrtf(s);
}

// Wt[c, r] = W[r, c]   (W: rows x cols)
__global__ void k_transpose(const float* W, float* Wt, int rows, int cols) {
  int i = blockIdx.x * blockDim.x + threadIdx.x;
  if (i >= rows * cols) return;
  int r = i / cols, c = i % cols;
  Wt[(long)c * rows + r] = W[i];
}

// ---------------------------------------------------------------- top-k select
// one wave32 per node row: coalesced lane-strided loads + shuffle reduction
__global__ void k_score(const float* __restrict__ X, const float* __restrict__ scorer,
                        const float* __restrict__ inv, const float* __restrict__ mask,
                        float* __restrict__ score, int Nn, int d) {
  int wave = (int)(((long)blockIdx.x * blockDim.x + threadIdx.x) >> 5);
  int lane = threadIdx.x & 31;
  if (wave >= Nn) return;
  const float* row = X + (long)wave * d;
  float s = 0.f;
  for (int i = lane; i < d; i += 32) s += row[i] * scorer[i];
  for (int o = 16; o; o >>= 1) s += __shfl_xor(s, o);
  if (lane == 0) score[wave] = s * (*inv) + mask[wave];
}

__device__ __forceinline__ unsigned score_key(float f) {
  unsigned u = __float_as_uint(f);
  return u ^ ((u >> 31) ? 0xFFFFFFFFu : 0x80000000u); // monotonic order-preserving key
}

// misc layout: [0..2047] histogram, 2048 thresh bin, 2049 count above, 2050/2051 counters
__global__ void k_hist(const float* __restrict__ score, int Nn, int* __restrict__ misc) {
  int n = blockIdx.x * blockDim.x + threadIdx.x;
  if (n >= Nn) return;
  atomicAdd(&misc[score_key(score[n]) >> 21], 1);
}

__global__ void k_thresh(int* misc, int k) {
  if (threadIdx.x != 0 || blockIdx.x != 0) return;
  int c = 0, b = 2047;
  for (; b > 0; --b) {
    int h = misc[b];
    if (c + h >= k) break;
    c += h;
  }
  misc[2048] = b;
  misc[2049] = c;
  misc[2050] = 0;
  misc[2051] = 0;
}

__global__ void k_collect(const float* __restrict__ score, int Nn, int* __restrict__ misc,
                          int* __restrict__ idx, int k) {
  int n = blockIdx.x * blockDim.x + threadIdx.x;
  if (n >= Nn) return;
  int bin = (int)(score_key(score[n]) >> 21);
  int tb = misc[2048];
  if (bin > tb) {
    int p = atomicAdd(&misc[2050], 1);
    if (p < k) idx[p] = n;
  } else if (bin == tb) {
    int p = misc[2049] + atomicAdd(&misc[2051], 1);
    if (p < k) idx[p] = n;
  }
}

// z[i,:] = X[idx[i],:] * tanh(score[idx[i]])
__global__ void k_gather_z(const float* __restrict__ X, const float* __restrict__ score,
                           const int* __restrict__ idx, float* __restrict__ z, int d) {
  int i = blockIdx.x, j = threadIdx.x;
  int n = idx[i];
  z[(long)i * d + j] = X[(long)n * d + j] * tanhf(score[n]);
}

// ---------------------------------------------------------------- layernorm / softmax
// one block (d=128 threads) per row; unbiased std (ddof=1) like torch/reference
__global__ void k_layernorm(const float* __restrict__ x, const float* __restrict__ a,
                            const float* __restrict__ b, float* __restrict__ y, int d) {
  int r = blockIdx.x, j = threadIdx.x;
  __shared__ float s1[128], s2[128];
  float v = x[(long)r * d + j];
  s1[j] = v;
  s2[j] = v * v;
  __syncthreads();
  for (int s = d >> 1; s > 0; s >>= 1) {
    if (j < s) { s1[j] += s1[j + s]; s2[j] += s2[j + s]; }
    __syncthreads();
  }
  float mu  = s1[0] / d;
  float var = (s2[0] - d * mu * mu) / (d - 1);
  float sd  = sqrtf(fmaxf(var, 0.f));
  y[(long)r * d + j] = a[j] * (v - mu) / (sd + LN_EPS) + b[j];
}

// row-wise softmax of S[L,L] (pre-scaled); block = L threads
__global__ void k_softmax(float* __restrict__ S, int L, float scale) {
  int r = blockIdx.x, j = threadIdx.x;
  __shared__ float red[128];
  float v = S[(long)r * L + j] * scale;
  red[j] = v;
  __syncthreads();
  for (int s = L >> 1; s > 0; s >>= 1) {
    if (j < s) red[j] = fmaxf(red[j], red[j + s]);
    __syncthreads();
  }
  float m = red[0];
  __syncthreads();
  float e = expf(v - m);
  red[j] = e;
  __syncthreads();
  for (int s = L >> 1; s > 0; s >>= 1) {
    if (j < s) red[j] += red[j + s];
    __syncthreads();
  }
  S[(long)r * L + j] = e / red[0];
}

// ---------------------------------------------------------------- WMMA GEMM
// C = A[M,K] @ B^T + bias (+resid) (+relu),  B stored [Nn,K] row-major.
// All weights are pre-transposed once per launch so this is the ONLY layout:
// every fragment load is a contiguous global_load_b128 run.
//   ctrans==0 : C[m,n] at C[m*Nn + n]   (row-major [M,Nn])
//   ctrans==1 : C[m,n] at C[n*M  + m]   (store transposed, for producing V^T)
// M multiple of 16, Nn multiple of 64, K multiple of 32.
// Each wave computes a 16x64 strip: 4 WMMAs reusing one A fragment (4x A reuse).
// fp32 -> f16 conversion at load (v_cvt_pk_f16_f32), f32 accumulation.
__global__ void k_gemm_wmma(const float* __restrict__ A, const float* __restrict__ B,
                            const float* __restrict__ bias, const float* __restrict__ resid,
                            float* __restrict__ C, int M, int Nn, int K,
                            int ctrans, int relu) {
  int lane  = threadIdx.x;               // 0..31
  int mTile = blockIdx.x * 4 + threadIdx.y;
  if (mTile >= (M >> 4)) return;          // uniform per wave: EXEC stays all-ones
  int m0 = mTile << 4;
  int n0 = blockIdx.y << 6;               // 64 output columns per wave
  int half = lane >> 4, lm = lane & 15;
  int h8 = half << 3;

  // A fragment rows: lane layout = two 8-contiguous-k runs at k0=h8 and k0=h8+16
  const float* Arow  = A + (long)(m0 + lm) * K + h8;
  const float* Brow0 = B + (long)(n0 + lm) * K + h8;

  v8f acc[4] = {{}, {}, {}, {}};

  for (int kb = 0; kb < K; kb += 32) {
    // speculative prefetch of the next A chunk (dropped if OOB)
    __builtin_prefetch(Arow + kb + 32, 0, 3);

    v16h af;
    {
      const float4 a0 = *(const float4*)(Arow + kb);
      const float4 a1 = *(const float4*)(Arow + kb + 4);
      const float4 a2 = *(const float4*)(Arow + kb + 16);
      const float4 a3 = *(const float4*)(Arow + kb + 20);
      af[0]  = (_Float16)a0.x; af[1]  = (_Float16)a0.y;
      af[2]  = (_Float16)a0.z; af[3]  = (_Float16)a0.w;
      af[4]  = (_Float16)a1.x; af[5]  = (_Float16)a1.y;
      af[6]  = (_Float16)a1.z; af[7]  = (_Float16)a1.w;
      af[8]  = (_Float16)a2.x; af[9]  = (_Float16)a2.y;
      af[10] = (_Float16)a2.z; af[11] = (_Float16)a2.w;
      af[12] = (_Float16)a3.x; af[13] = (_Float16)a3.y;
      af[14] = (_Float16)a3.z; af[15] = (_Float16)a3.w;
    }

#pragma unroll
    for (int nt = 0; nt < 4; ++nt) {
      const float* Brow = Brow0 + (long)(nt << 4) * K + kb;
      v16h bf;
      const float4 b0 = *(const float4*)(Brow);
      const float4 b1 = *(const float4*)(Brow + 4);
      const float4 b2 = *(const float4*)(Brow + 16);
      const float4 b3 = *(const float4*)(Brow + 20);
      bf[0]  = (_Float16)b0.x; bf[1]  = (_Float16)b0.y;
      bf[2]  = (_Float16)b0.z; bf[3]  = (_Float16)b0.w;
      bf[4]  = (_Float16)b1.x; bf[5]  = (_Float16)b1.y;
      bf[6]  = (_Float16)b1.z; bf[7]  = (_Float16)b1.w;
      bf[8]  = (_Float16)b2.x; bf[9]  = (_Float16)b2.y;
      bf[10] = (_Float16)b2.z; bf[11] = (_Float16)b2.w;
      bf[12] = (_Float16)b3.x; bf[13] = (_Float16)b3.y;
      bf[14] = (_Float16)b3.z; bf[15] = (_Float16)b3.w;
      acc[nt] = __builtin_amdgcn_wmma_f32_16x16x32_f16(false, af, false, bf,
                                                       (short)0, acc[nt], false, false);
    }
  }

  // D layout: VGPR j holds row m = 8*half + j, lane gives column n = lm
#pragma unroll
  for (int nt = 0; nt < 4; ++nt) {
    int n = n0 + (nt << 4) + lm;
    float bv = bias ? bias[n] : 0.f;
#pragma unroll
    for (int j = 0; j < 8; ++j) {
      int m = m0 + h8 + j;
      float v = acc[nt][j] + bv;
      if (resid) v += resid[(long)m * Nn + n];
      if (relu)  v = fmaxf(v, 0.f);
      if (ctrans) C[(long)n * M + m] = v;   // store C^T (leading dim M)
      else        C[(long)m * Nn + n] = v;
    }
  }
}

// ---------------------------------------------------------------- edge scatter-add
// out[dst[e], :] += XW[src[e], :] * w[e]   (one wave32 per edge; atomics RMW in L2)
__global__ void k_scatter(const float* __restrict__ XW, const int* __restrict__ src,
                          const int* __restrict__ dst, const float* __restrict__ wgt,
                          float* __restrict__ out, int E, int dout) {
  long t = (long)blockIdx.x * blockDim.x + threadIdx.x;
  int e = (int)(t >> 5);
  if (e >= E) return;
  int lane = (int)(t & 31);
  int s = src[e], dd = dst[e];
  float we = wgt[e];
  const float* xr = XW + (long)s * dout;
  float* orow = out + (long)dd * dout;
  for (int j = lane; j < dout; j += 32)
    __hip_atomic_fetch_add(&orow[j], xr[j] * we, __ATOMIC_RELAXED, __HIP_MEMORY_SCOPE_AGENT);
}

// ---------------------------------------------------------------- host orchestration
struct LP {                 // raw (untransposed) parameter pointers
  const float* scorer;
  const float* W0;
  const float* aw[8];       // a1_q,a1_k,a1_v,a1_o,a2_q,a2_k,a2_v,a2_o  [128,128]
  const float* ab[8];       // biases [128]
  const float *ff1w, *ff1b, *ff2w, *ff2b;
  const float* nrm[6];      // n1a,n1b,n2a,n2b,n3a,n3b
};

struct LPT {                // per-launch transposed weights (workspace)
  float* awt[8];            // [128,128] transposed
  float* ff1wt;             // [2048,128]
  float* ff2wt;             // [128,2048]
};

static LP make_lp(void* const* in, int base) {
  LP p;
  p.scorer = (const float*)in[base + 0];
  p.W0     = (const float*)in[base + 1];
  for (int i = 0; i < 8; ++i) {
    p.aw[i] = (const float*)in[base + 2 + 2 * i];
    p.ab[i] = (const float*)in[base + 3 + 2 * i];
  }
  p.ff1w = (const float*)in[base + 18];
  p.ff1b = (const float*)in[base + 19];
  p.ff2w = (const float*)in[base + 20];
  p.ff2b = (const float*)in[base + 21];
  for (int j = 0; j < 6; ++j) p.nrm[j] = (const float*)in[base + 22 + j];
  return p;
}

struct Scr {
  float *XW, *z, *x2, *Q, *K, *Vt, *S, *A, *ff1, *score;
  int *misc, *topk;
};

static inline void gemm(const float* A, const float* B, const float* bias, const float* resid,
                        float* C, int M, int Nn, int K, int ctrans, int relu, hipStream_t st) {
  dim3 blk(32, 4);
  dim3 grd(((unsigned)(M >> 4) + 3) >> 2, (unsigned)(Nn >> 6));
  k_gemm_wmma<<<grd, blk, 0, st>>>(A, B, bias, resid, C, M, Nn, K, ctrans, relu);
}

// x (= W^T, [L,128]) evolved in place by the decoder cell; z: [L,128] memory
static void run_decoder(float* x, const float* z, int L, const LP& P, const LPT& W,
                        Scr& S, hipStream_t st) {
  const int d = DIN;
  const float scale = 0.08838834764831845f;  // 1/sqrt(128)
  for (int blkI = 0; blkI < 2; ++blkI) {
    int o = blkI * 4;
    k_layernorm<<<L, d, 0, st>>>(x, P.nrm[2 * blkI], P.nrm[2 * blkI + 1], S.x2, d);
    const float* kv = (blkI == 0) ? S.x2 : z;   // self-attn then cross-attn on top-k feats
    gemm(S.x2, W.awt[o + 0], P.ab[o + 0], nullptr, S.Q,  L, d, d, 0, 0, st);
    gemm(kv,   W.awt[o + 1], P.ab[o + 1], nullptr, S.K,  L, d, d, 0, 0, st);
    gemm(kv,   W.awt[o + 2], P.ab[o + 2], nullptr, S.Vt, L, d, d, 1, 0, st); // store V^T [d,L]
    gemm(S.Q, S.K, nullptr, nullptr, S.S, L, L, d, 0, 0, st);   // Q @ K^T  (K is [L,d]=[Nn,K])
    k_softmax<<<L, L, 0, st>>>(S.S, L, scale);
    gemm(S.S, S.Vt, nullptr, nullptr, S.A, L, d, L, 0, 0, st);  // S @ V    (V^T is [d,L]=[Nn,K])
    gemm(S.A, W.awt[o + 3], P.ab[o + 3], x, x, L, d, d, 0, 0, st); // x += attn @ Wo + bo
  }
  k_layernorm<<<L, d, 0, st>>>(x, P.nrm[4], P.nrm[5], S.x2, d);
  gemm(S.x2, W.ff1wt, P.ff1b, nullptr, S.ff1, L, DFF, d, 0, 1, st);  // relu(x2 W1 + b1)
  gemm(S.ff1, W.ff2wt, P.ff2b, x, x, L, d, DFF, 0, 0, st);           // x += ff W2 + b2
}

static void layer_step(const float* X, const LP& P, const LPT& W, float* Wt, int dout,
                       const float* mask_t, const int* src_t, const int* dst_t,
                       const float* w_t, const float* invn, float* hout,
                       Scr& S, hipStream_t st) {
  const int d = DIN;
  int nb = (N_NODES + 255) / 256;
  int nbw = (int)(((long)N_NODES * 32 + 255) / 256);   // wave-per-row grids
  // --- top-k node selection + z features (radix-threshold select) ---
  k_zero_i32<<<9, 256, 0, st>>>(S.misc, 2052);
  k_score  <<<nbw, 256, 0, st>>>(X, P.scorer, invn, mask_t, S.score, N_NODES, d);
  k_hist   <<<nb, 256, 0, st>>>(S.score, N_NODES, S.misc);
  k_thresh <<<1, 1, 0, st>>>(S.misc, dout);
  k_collect<<<nb, 256, 0, st>>>(S.score, N_NODES, S.misc, S.topk, dout);
  k_gather_z<<<dout, d, 0, st>>>(X, S.score, S.topk, S.z, d);
  // --- evolve weights (Wt holds W^T [dout,128]) ---
  run_decoder(Wt, S.z, dout, P, W, S, st);
  // --- XW = X @ W  (== X @ Wt^T; Wt is [dout,128]=[Nn,K]) ---
  gemm(X, Wt, nullptr, nullptr, S.XW, N_NODES, dout, d, 0, 0, st);
  // --- sparse Ahat @ XW via edge scatter-add, then ReLU ---
  long hn = (long)N_NODES * dout;
  k_zero_f32<<<4096, 256, 0, st>>>(hout, hn);
  long tt = (long)E_EDGES * 32;
  k_scatter<<<(int)((tt + 255) / 256), 256, 0, st>>>(S.XW, src_t, dst_t, w_t, hout,
                                                     E_EDGES, dout);
  k_relu<<<4096, 256, 0, st>>>(hout, hn);
}

static void transpose_weights(const LP& P, LPT& W, hipStream_t st) {
  for (int i = 0; i < 8; ++i)
    k_transpose<<<(128 * 128 + 255) / 256, 256, 0, st>>>(P.aw[i], W.awt[i], 128, 128);
  k_transpose<<<(128 * DFF + 255) / 256, 256, 0, st>>>(P.ff1w, W.ff1wt, 128, DFF);
  k_transpose<<<(DFF * 128 + 255) / 256, 256, 0, st>>>(P.ff2w, W.ff2wt, DFF, 128);
}

extern "C" void kernel_launch(void* const* d_in, const int* in_sizes, int n_in,
                              void* d_out, int out_size, void* d_ws, size_t ws_size,
                              hipStream_t stream) {
  (void)in_sizes; (void)n_in; (void)out_size; (void)ws_size;
  const float* x_seq = (const float*)d_in[0];
  const float* mask  = (const float*)d_in[1];
  const float* wgt   = (const float*)d_in[2];
  const int*   src   = (const int*)d_in[3];
  const int*   dst   = (const int*)d_in[4];
  LP P1 = make_lp(d_in, 5);    // params1 leaves in dict insertion order
  LP P2 = make_lp(d_in, 33);   // params2
  float* out = (float*)d_out;  // [N, 64] fp32

  // workspace carve (~60 MB): h1 + XW are L2-resident hot buffers
  float* p = (float*)d_ws;
  Scr S;
  float* h1 = p;  p += (long)N_NODES * 128;
  S.XW  = p;      p += (long)N_NODES * 128;
  float* Wt1 = p; p += 128 * 128;
  float* Wt2 = p; p += 64 * 128;
  S.z   = p;      p += 128 * 128;
  S.x2  = p;      p += 128 * 128;
  S.Q   = p;      p += 128 * 128;
  S.K   = p;      p += 128 * 128;
  S.Vt  = p;      p += 128 * 128;
  S.S   = p;      p += 128 * 128;
  S.A   = p;      p += 128 * 128;
  S.ff1 = p;      p += 128 * DFF;
  LPT W1, W2;
  for (int i = 0; i < 8; ++i) { W1.awt[i] = p; p += 128 * 128; }
  W1.ff1wt = p;   p += 128 * DFF;
  W1.ff2wt = p;   p += DFF * 128;
  for (int i = 0; i < 8; ++i) { W2.awt[i] = p; p += 128 * 128; }
  W2.ff1wt = p;   p += 128 * DFF;
  W2.ff2wt = p;   p += DFF * 128;
  S.score = p;    p += N_NODES;
  float* inv1 = p; p += 1;
  float* inv2 = p; p += 1;
  S.misc = (int*)p; p += 2052;
  S.topk = (int*)p; p += 128;

  k_inv_norm<<<1, 32, 0, stream>>>(P1.scorer, 128, inv1);
  k_inv_norm<<<1, 32, 0, stream>>>(P2.scorer, 128, inv2);
  // evolving weights kept transposed: Wt = W0^T, so decoder acts on it directly
  k_transpose<<<(128 * 128 + 255) / 256, 256, 0, stream>>>(P1.W0, Wt1, 128, 128);
  k_transpose<<<(128 * 64 + 255) / 256, 256, 0, stream>>>(P2.W0, Wt2, 128, 64);
  // pre-transpose all static decoder weights -> every GEMM B-operand is [N,K]
  transpose_weights(P1, W1, stream);
  transpose_weights(P2, W2, stream);

  // Interleave the two GRCU scans: layer2 step t only needs h1[t], so a single
  // [N,128] h1 buffer suffices and stays resident in the 192 MB L2.
  for (int t = 0; t < T_STEPS; ++t) {
    const float* Xt  = x_seq + (long)t * N_NODES * 128;
    const float* mt  = mask  + (long)t * N_NODES;
    const int*   st_ = src   + (long)t * E_EDGES;
    const int*   dt_ = dst   + (long)t * E_EDGES;
    const float* wt_ = wgt   + (long)t * E_EDGES;
    layer_step(Xt, P1, W1, Wt1, 128, mt, st_, dt_, wt_, inv1, h1, S, stream);
    layer_step(h1, P2, W2, Wt2, 64,  mt, st_, dt_, wt_, inv2, out, S, stream);
  }
}
